// A3M_32590211842278
// MI455X (gfx1250) — compile-verified, matrix operand order checked by reading, not run on hardware
//
#include <hip/hip_runtime.h>
#include <hip/hip_bf16.h>
#include <math.h>

#define DD   768
#define BB   256
#define LL   81
#define VV   49
#define MM   16
#define NROW (BB * MM)      // 4096 rows of noun_embed
#define KT   (DD / 32)      // 24 k-steps (bf16 wmma K=32)
#define NT   (DD / 16)      // 48 n-tiles
#define NWAVE 8
#define NTPW (NT / NWAVE)   // 6 n-tiles per wave

typedef __attribute__((ext_vector_type(16))) __bf16 v16bf;
typedef __attribute__((ext_vector_type(8)))  __bf16 v8bf;
typedef __attribute__((ext_vector_type(8)))  float  v8f;

// ---------------- prep: WnT (bf16 transpose of Wn) ----------------
__global__ void A3M_prep_wnt(const float* __restrict__ Wn, __bf16* __restrict__ WnT) {
    int idx = blockIdx.x * blockDim.x + threadIdx.x;   // over 768*768, k-major read
    if (idx < DD * DD) {
        int k = idx / DD, n = idx % DD;
        WnT[(size_t)n * DD + k] = (__bf16)Wn[idx];
    }
}

// ---------------- prep: u1 = Wa1@Wl[:D], u2 = Wa2@Wl[D:], c ----------------
__global__ void A3M_prep_u(const float* __restrict__ Wa1, const float* __restrict__ Wa2,
                           const float* __restrict__ Wl,  const float* __restrict__ ba1,
                           const float* __restrict__ ba2, const float* __restrict__ bl,
                           float* __restrict__ u1, float* __restrict__ u2,
                           float* __restrict__ cbase) {
    int k = blockIdx.x * blockDim.x + threadIdx.x;
    if (k < DD) {
        float s1 = 0.f, s2 = 0.f;
        for (int j = 0; j < DD; ++j) {
            s1 += Wa1[(size_t)k * DD + j] * Wl[j];
            s2 += Wa2[(size_t)k * DD + j] * Wl[DD + j];
        }
        u1[k] = s1;
        u2[k] = s2;
    }
    if (blockIdx.x == 0 && threadIdx.x == 0) {
        float c = bl[0];
        for (int j = 0; j < DD; ++j) c += ba1[j] * Wl[j] + ba2[j] * Wl[DD + j];
        *cbase = c;
    }
}

// ---------------- gather noun_embed (f32 + bf16) ----------------
__global__ void A3M_gather(const float* __restrict__ token, const float* __restrict__ video,
                           const int* __restrict__ masks,
                           float* __restrict__ nounF, __bf16* __restrict__ nounB) {
    __shared__ int posS[MM];
    const int b = blockIdx.x;
    if (threadIdx.x == 0) {
        int cnt = 0;
        for (int l = 0; l < LL && cnt < MM; ++l)
            if (masks[b * LL + l]) posS[cnt++] = l;
        for (; cnt < MM; ++cnt) posS[cnt] = 0;   // pad gathers position 0
    }
    __syncthreads();
    const size_t base = (size_t)b * MM * DD;
    for (int idx = threadIdx.x; idx < MM * DD; idx += blockDim.x) {
        int m = idx / DD, d = idx - m * DD;
        int l = posS[m];
        float val = (l >= 1 && l < VV) ? video[((size_t)b * VV + l) * DD + d]
                                       : token[((size_t)b * LL + l) * DD + d];
        nounF[base + idx] = val;
        nounB[base + idx] = (__bf16)val;
    }
}

// ---------------- WMMA GEMM fused: s_n = tanh(noun@Wn + bn) @ Wa[:D] ----------------
// kt is the OUTER loop: one A-fragment (2 ds_load_b128) reused across the wave's
// 6 N-tiles, which own 6 persistent v8f accumulators. Avoids the A-hoist spill.
__global__ void A3M_sn_gemm(const __bf16* __restrict__ A,   // [4096, 768] noun bf16
                            const __bf16* __restrict__ BT,  // WnT [n][k] bf16
                            const float* __restrict__ bn,
                            const float* __restrict__ Wa,   // first 768 entries used
                            float* __restrict__ s_n) {
    __shared__ __bf16 As[MM * DD];
    __shared__ float sred[MM];
    const int tid = threadIdx.x;
    const int rowBase = blockIdx.x * MM;

    // stage 16x768 A tile into LDS (16B chunks)
    {
        const v8bf* src = (const v8bf*)(A + (size_t)rowBase * DD);
        v8bf* dst = (v8bf*)As;
        for (int i = tid; i < MM * DD / 8; i += blockDim.x) dst[i] = src[i];
    }
    if (tid < MM) sred[tid] = 0.f;
    __syncthreads();

    const int wave   = tid >> 5;
    const int lane   = tid & 31;
    const int lane16 = lane & 15;
    const int half   = lane >> 4;

    v8f c[NTPW];
#pragma unroll
    for (int i = 0; i < NTPW; ++i) c[i] = v8f{};

    const __bf16* asrow = As + lane16 * DD + half * 8;
    // B row base for this lane within each of the wave's 6 n-tiles
    const __bf16* brow0 = BT + (size_t)(wave * 16 + lane16) * DD + half * 8;

#pragma unroll 1
    for (int kt = 0; kt < KT; ++kt) {
        const int k0 = kt * 32;
        union { v16bf v; v8bf h[2]; } ua;
        ua.h[0] = *(const v8bf*)(asrow + k0);
        ua.h[1] = *(const v8bf*)(asrow + k0 + 16);
#pragma unroll
        for (int i = 0; i < NTPW; ++i) {
            const __bf16* brow = brow0 + (size_t)(i * 8 * 16) * DD + k0;
            union { v16bf v; v8bf h[2]; } ub;
            ub.h[0] = *(const v8bf*)(brow);
            ub.h[1] = *(const v8bf*)(brow + 16);
            c[i] = __builtin_amdgcn_wmma_f32_16x16x32_bf16(
                       false, ua.v, false, ub.v, (short)0, c[i], false, false);
        }
    }

    // epilogue: lane holds n = nt*16 + lane16 ; vgpr r -> m = r + 8*half
    float acc[8];
#pragma unroll
    for (int r = 0; r < 8; ++r) acc[r] = 0.f;
#pragma unroll
    for (int i = 0; i < NTPW; ++i) {
        const int   n   = (wave + 8 * i) * 16 + lane16;
        const float bnn = bn[n];
        const float wan = Wa[n];
#pragma unroll
        for (int r = 0; r < 8; ++r) acc[r] += tanhf(c[i][r] + bnn) * wan;
    }

    // reduce over the 16 lanes of each half (xor of bits 0..3 stays in-half)
#pragma unroll
    for (int r = 0; r < 8; ++r) {
        float s = acc[r];
        s += __shfl_xor(s, 1, 32);
        s += __shfl_xor(s, 2, 32);
        s += __shfl_xor(s, 4, 32);
        s += __shfl_xor(s, 8, 32);
        acc[r] = s;
    }
    if (lane16 == 0) {
#pragma unroll
        for (int r = 0; r < 8; ++r) atomicAdd(&sred[r + 8 * half], acc[r]);
    }
    __syncthreads();
    if (tid < MM) s_n[rowBase + tid] = sred[tid];
}

// ---------------- softmax(s_n) -> v[b,:], abase[b] = v.u2 + c ----------------
__global__ void A3M_softmax_v(const float* __restrict__ s_n, const float* __restrict__ nounF,
                              const float* __restrict__ u2,  const float* __restrict__ cbase,
                              float* __restrict__ v, float* __restrict__ abase) {
    __shared__ float p[MM];
    __shared__ float dotred;
    const int b = blockIdx.x;
    if (threadIdx.x == 0) {
        float mx = -1e30f;
        for (int m = 0; m < MM; ++m) mx = fmaxf(mx, s_n[b * MM + m]);
        float sum = 0.f;
        for (int m = 0; m < MM; ++m) { float e = __expf(s_n[b * MM + m] - mx); p[m] = e; sum += e; }
        float inv = 1.f / sum;
        for (int m = 0; m < MM; ++m) p[m] *= inv;
        dotred = 0.f;
    }
    __syncthreads();
    float part = 0.f;
    for (int d = threadIdx.x; d < DD; d += blockDim.x) {
        float vd = 0.f;
#pragma unroll
        for (int m = 0; m < MM; ++m) vd += p[m] * nounF[((size_t)b * MM + m) * DD + d];
        v[(size_t)b * DD + d] = vd;
        part += vd * u2[d];
    }
    atomicAdd(&dotred, part);
    __syncthreads();
    if (threadIdx.x == 0) abase[b] = dotred + *cbase;
}

// ---------------- fused gate + output: one wave per (b,l) row ----------------
__global__ void A3M_fuse_out(const float* __restrict__ token, const float* __restrict__ video,
                             const float* __restrict__ u1, const float* __restrict__ v,
                             const float* __restrict__ abase, float* __restrict__ out) {
    const int wave = threadIdx.x >> 5;
    const int lane = threadIdx.x & 31;
    const int row  = blockIdx.x * 8 + wave;       // 2592*8 == 20736 exactly
    const int b = row / LL, l = row - b * LL;
    const float* src = (l >= 1 && l < VV) ? video + ((size_t)b * VV + l) * DD
                                          : token + ((size_t)b * LL + l) * DD;
    float r[24];
    float dot = 0.f;
#pragma unroll
    for (int i = 0; i < 24; ++i) {
        int d = lane + i * 32;
        r[i] = src[d];
        dot += r[i] * u1[d];
    }
    dot += __shfl_xor(dot, 1, 32);
    dot += __shfl_xor(dot, 2, 32);
    dot += __shfl_xor(dot, 4, 32);
    dot += __shfl_xor(dot, 8, 32);
    dot += __shfl_xor(dot, 16, 32);
    const float alpha = 1.f / (1.f + __expf(-(dot + abase[b])));
    const float* vb = v + (size_t)b * DD;
    float* orow = out + (size_t)row * DD;
#pragma unroll
    for (int i = 0; i < 24; ++i) {
        int d = lane + i * 32;
        orow[d] = (1.f - alpha) * r[i] + alpha * vb[d];
    }
}

// ---------------- launcher ----------------
extern "C" void kernel_launch(void* const* d_in, const int* in_sizes, int n_in,
                              void* d_out, int out_size, void* d_ws, size_t ws_size,
                              hipStream_t stream) {
    const float* token = (const float*)d_in[0];
    const float* video = (const float*)d_in[1];
    const int*   masks = (const int*)d_in[2];
    const float* Wn    = (const float*)d_in[4];
    const float* bn    = (const float*)d_in[5];
    const float* Wa    = (const float*)d_in[8];   // [1536]; first 768 used (noun side)
    const float* Wl    = (const float*)d_in[10];  // [1536]
    const float* bl    = (const float*)d_in[11];
    const float* Wa1   = (const float*)d_in[12];
    const float* ba1   = (const float*)d_in[13];
    const float* Wa2   = (const float*)d_in[14];
    const float* ba2   = (const float*)d_in[15];
    float* out = (float*)d_out;

    // workspace layout (256B aligned)
    char* ws = (char*)d_ws;
    size_t off = 0;
    auto alloc = [&](size_t bytes) { void* p = ws + off; off = (off + bytes + 255) & ~(size_t)255; return p; };
    __bf16* WnT   = (__bf16*)alloc((size_t)DD * DD * 2);       // 1.18 MB
    __bf16* nounB = (__bf16*)alloc((size_t)NROW * DD * 2);     // 6.3 MB
    float*  nounF = (float*) alloc((size_t)NROW * DD * 4);     // 12.6 MB
    float*  u1    = (float*) alloc(DD * 4);
    float*  u2    = (float*) alloc(DD * 4);
    float*  cbase = (float*) alloc(4);
    float*  s_n   = (float*) alloc(NROW * 4);
    float*  v     = (float*) alloc((size_t)BB * DD * 4);       // 0.79 MB
    float*  abase = (float*) alloc(BB * 4);
    (void)ws_size; (void)n_in; (void)in_sizes; (void)out_size;

    A3M_prep_wnt <<<(DD * DD + 255) / 256, 256, 0, stream>>>(Wn, WnT);
    A3M_prep_u   <<<(DD + 255) / 256, 256, 0, stream>>>(Wa1, Wa2, Wl, ba1, ba2, bl, u1, u2, cbase);
    A3M_gather   <<<BB, 256, 0, stream>>>(token, video, masks, nounF, nounB);
    A3M_sn_gemm  <<<NROW / MM, 256, 0, stream>>>(nounB, WnT, bn, Wa, s_n);
    A3M_softmax_v<<<BB, 256, 0, stream>>>(s_n, nounF, u2, cbase, v, abase);
    A3M_fuse_out <<<(BB * LL) / 8, 256, 0, stream>>>(token, video, u1, v, abase, out);
}
